// InstanceEmbeddingLoss_730144440790
// MI455X (gfx1250) — compile-verified, matrix operand order checked by reading, not run on hardware
//
#include <hip/hip_runtime.h>
#include <hip/hip_bf16.h>
#include <stdint.h>

typedef __attribute__((ext_vector_type(16))) _Float16 v16h;
typedef __attribute__((ext_vector_type(8)))  float    v8f;
typedef unsigned int  u32;
typedef unsigned char u8;

#define HW    192
#define NPIX  (HW*HW)            // 36864
#define C_IN  16
#define HID   32
#define NINST 32
#define PADW  256
#define PPIX  (PADW*PADW)        // 65536 padded pixels
#define PTOT  (NINST*PPIX)       // 2097152 elements per image
#define BLKE  4096               // elements per sort block
#define NBLK  (PTOT/BLKE)        // 512 sort blocks
#define EPSF  1e-6f

// ---- monotone float <-> uint key transform; radix-ascending == error-descending ----
__device__ __forceinline__ u32 enc_key(float f){
  u32 u = __float_as_uint(f);
  u32 a = (u & 0x80000000u) ? ~u : (u | 0x80000000u);  // ascending order
  return ~a;                                           // flip for descending
}
__device__ __forceinline__ float dec_key(u32 key){
  u32 a = ~key;
  u32 u = (a & 0x80000000u) ? (a ^ 0x80000000u) : ~a;
  return __uint_as_float(u);
}

__global__ void k_zero(float* p, int n){
  int i = blockIdx.x*blockDim.x + threadIdx.x;
  if (i < n) p[i] = 0.f;
}

// ---- Stage 1: deterministic weighted centroid accumulation, one block per (image, instance)
__global__ void k_stage1(const float* __restrict__ emb, const float* __restrict__ w,
                         const int* __restrict__ gt,
                         float* __restrict__ counts_acc, float* __restrict__ centers_acc,
                         float* __restrict__ gts){
  __shared__ float red[256];
  int tid = threadIdx.x;
  int n = blockIdx.x, b = blockIdx.y;
  float accC[C_IN];
  #pragma unroll
  for (int c = 0; c < C_IN; ++c) accC[c] = 0.f;
  float accW = 0.f, accN = 0.f;
  for (int pix = tid; pix < NPIX; pix += 256){
    if (gt[b*NPIX + pix] == n + 1){
      float wv = w[b*NPIX + pix];
      accW += wv; accN += 1.f;
      #pragma unroll
      for (int c = 0; c < C_IN; ++c)
        accC[c] = fmaf(emb[(b*C_IN + c)*NPIX + pix], wv, accC[c]);
    }
  }
  for (int comp = 0; comp < C_IN + 2; ++comp){
    float v = (comp < C_IN) ? accC[comp] : (comp == C_IN ? accW : accN);
    red[tid] = v; __syncthreads();
    for (int off = 128; off > 0; off >>= 1){
      if (tid < off) red[tid] += red[tid + off];
      __syncthreads();
    }
    if (tid == 0){
      if (comp < C_IN)       centers_acc[(b*NINST + n)*C_IN + comp] = red[0];
      else if (comp == C_IN) counts_acc[b*NINST + n] = red[0];
      else                   atomicAdd(&gts[b], red[0]);  // exact integer-valued
    }
    __syncthreads();
  }
}

__global__ void k_fincenters(const float* __restrict__ centers_acc,
                             const float* __restrict__ counts_acc,
                             float* __restrict__ centers){
  int i = blockIdx.x*blockDim.x + threadIdx.x;
  if (i < 4*NINST*C_IN){
    int b = i >> 9;
    int n = (i >> 4) & 31;
    centers[i] = centers_acc[i] / (counts_acc[b*NINST + n] + EPSF);
  }
}

// A1b[b][n][k] = b1[k] + sum_c centers[b][n][c] * W1[c][k]
__global__ void k_a1b(const float* __restrict__ centers, const float* __restrict__ W1,
                      const float* __restrict__ b1, float* __restrict__ A1b){
  int i = blockIdx.x*blockDim.x + threadIdx.x;
  if (i < 4*NINST*HID){
    int b = i >> 10;
    int n = (i >> 5) & 31;
    int k = i & 31;
    float acc = b1[k];
    #pragma unroll
    for (int c = 0; c < C_IN; ++c)
      acc = fmaf(centers[b*NINST*C_IN + n*C_IN + c], W1[c*HID + k], acc);
    A1b[i] = acc;
  }
}

// ---- Stage 2: E1[p][k] = sum_c embPadded[c][p] * W1[c][k] via v_wmma_f32_16x16x32_f16
// One wave per 16-pixel M tile; two N tiles (hidden 0..15, 16..31); K=channels, 16..31 zero.
__global__ void k_e1_wmma(const float* __restrict__ emb, const float* __restrict__ W1,
                          float* __restrict__ E1, int b){
  int lane = threadIdx.x & 31;
  int wv   = threadIdx.x >> 5;
  int tile = blockIdx.x*8 + wv;
  int pixelBase = tile << 4;
  int p = pixelBase + (lane & 15);        // A-matrix row M = lane % 16
  int y = p >> 8, x = p & 255;
  bool inb = (y >= 32 && y < 32 + HW && x >= 32 && x < 32 + HW);
  int cBase = (lane < 16) ? 0 : 8;        // K-half per lane group (ISA 16-bit A layout)
  v16h a;
  #pragma unroll
  for (int j = 0; j < 8; ++j){
    float v = 0.f;
    if (inb) v = emb[((b*C_IN + (cBase + j))*HW + (y - 32))*HW + (x - 32)];
    a[j]     = (_Float16)v;               // K = cBase..cBase+7
    a[8 + j] = (_Float16)0.f;             // K = 16..31 zero pad
  }
  int ncol = lane & 15;                   // B/D column N = lane % 16
  #pragma unroll
  for (int t = 0; t < 2; ++t){
    v16h bv;
    #pragma unroll
    for (int j = 0; j < 8; ++j){
      bv[j]     = (_Float16)W1[(cBase + j)*HID + t*16 + ncol];
      bv[8 + j] = (_Float16)0.f;
    }
    v8f c = {0.f,0.f,0.f,0.f,0.f,0.f,0.f,0.f};
    v8f d = __builtin_amdgcn_wmma_f32_16x16x32_f16(
        false, a, false, bv, (short)0, c, false, false);
    #pragma unroll
    for (int q = 0; q < 8; ++q){
      int m = q + ((lane >> 4) << 3);     // D row per ISA C/D layout
      E1[(size_t)(pixelBase + m)*HID + t*16 + ncol] = d[q];
    }
  }
}

// ---- Stage 3: per (instance, pixel) logits -> error keys + label payloads
__global__ void k_errors(const float* __restrict__ E1, const float* __restrict__ A1b,
                         const float* __restrict__ W2, const float* __restrict__ b2,
                         const int* __restrict__ gt, u32* __restrict__ keys,
                         u8* __restrict__ vals, int b){
  __shared__ float sA[NINST*HID];
  __shared__ float sW2[HID];
  int tid = threadIdx.x;
  for (int i = tid; i < NINST*HID; i += blockDim.x) sA[i] = A1b[b*NINST*HID + i];
  if (tid < HID) sW2[tid] = W2[tid];
  __syncthreads();
  int p = blockIdx.x*blockDim.x + tid;
  float B2 = b2[0];
  float e1[HID];
  const float4* E4 = (const float4*)(E1 + (size_t)p*HID);
  #pragma unroll
  for (int q = 0; q < HID/4; ++q){
    float4 f = E4[q];
    e1[4*q+0]=f.x; e1[4*q+1]=f.y; e1[4*q+2]=f.z; e1[4*q+3]=f.w;
  }
  int y = p >> 8, x = p & 255;
  int g = 0;
  if (y >= 32 && y < 32 + HW && x >= 32 && x < 32 + HW)
    g = gt[b*NPIX + (y - 32)*HW + (x - 32)];
  for (int n = 0; n < NINST; ++n){
    float acc = B2;
    #pragma unroll
    for (int k = 0; k < HID; ++k){
      float h = sA[n*HID + k] - e1[k];
      acc = fmaf(fmaxf(h, 0.f), sW2[k], acc);
    }
    int lab = (g == n + 1) ? 1 : 0;
    float sgn = lab ? 1.f : -1.f;
    float err = 1.f - acc*sgn;
    size_t idx = (size_t)n*PPIX + p;     // matches reshape(-1) order
    keys[idx] = enc_key(err);
    vals[idx] = (u8)lab;
  }
}

// ---- Radix sort: 8 passes of 4 bits, stable, key + 1-byte label payload ----
__global__ void k_hist(const u32* __restrict__ keys, u32* __restrict__ hist, int shift){
  __shared__ u32 sh[16];
  int tid = threadIdx.x;
  if (tid < 16) sh[tid] = 0;
  __syncthreads();
  int base = blockIdx.x*BLKE + tid*16;
  #pragma unroll
  for (int j = 0; j < 16; ++j){
    u32 d = (keys[base + j] >> shift) & 15u;
    atomicAdd(&sh[d], 1u);
  }
  __syncthreads();
  if (tid < 16) hist[tid*NBLK + blockIdx.x] = sh[tid];   // digit-major
}

// single-block exclusive scan, n multiple of 256
__global__ void k_scan_excl(u32* data, int n){
  __shared__ u32 s[256];
  int tid = threadIdx.x;
  u32 carry = 0;
  for (int base = 0; base < n; base += 256){
    u32 v = data[base + tid];
    s[tid] = v; __syncthreads();
    for (int off = 1; off < 256; off <<= 1){
      u32 xv = (tid >= off) ? s[tid - off] : 0u;
      __syncthreads();
      s[tid] += xv;
      __syncthreads();
    }
    data[base + tid] = s[tid] - v + carry;
    u32 tot = s[255];
    __syncthreads();
    carry += tot;
  }
}

__global__ void k_scatter(const u32* __restrict__ keysIn, const u8* __restrict__ valsIn,
                          u32* __restrict__ keysOut, u8* __restrict__ valsOut,
                          const u32* __restrict__ hist, int shift){
  __shared__ u32 cnt[16*256];
  int tid = threadIdx.x;
  int base = blockIdx.x*BLKE + tid*16;   // thread-blocked => stable
  u32 k[16]; u8 v[16];
  #pragma unroll
  for (int j = 0; j < 16; ++j){ k[j] = keysIn[base + j]; v[j] = valsIn[base + j]; }
  #pragma unroll
  for (int d = 0; d < 16; ++d){
    u32 c = 0;
    #pragma unroll
    for (int j = 0; j < 16; ++j) c += (((k[j] >> shift) & 15u) == (u32)d);
    cnt[d*256 + tid] = c;
  }
  __syncthreads();
  if (tid < 16){                          // exclusive scan of each digit row
    u32 run = 0;
    for (int i = 0; i < 256; ++i){
      u32 t = cnt[tid*256 + i];
      cnt[tid*256 + i] = run;
      run += t;
    }
  }
  __syncthreads();
  #pragma unroll
  for (int j = 0; j < 16; ++j){
    u32 d = (k[j] >> shift) & 15u;
    u32 r = 0;
    #pragma unroll
    for (int i = 0; i < 16; ++i) if (i < j) r += (((k[i] >> shift) & 15u) == d);
    u32 dst = hist[d*NBLK + blockIdx.x] + cnt[d*256 + tid] + r;
    keysOut[dst] = k[j];
    valsOut[dst] = v[j];
  }
}

// ---- Label prefix sums + closed-form Lovasz gradient dot product ----
__global__ void k_bsum(const u8* __restrict__ vals, u32* __restrict__ bsum){
  __shared__ u32 s[256];
  int tid = threadIdx.x;
  int base = blockIdx.x*BLKE + tid*16;
  u32 t = 0;
  #pragma unroll
  for (int j = 0; j < 16; ++j) t += vals[base + j];
  s[tid] = t; __syncthreads();
  for (int off = 128; off > 0; off >>= 1){
    if (tid < off) s[tid] += s[tid + off];
    __syncthreads();
  }
  if (tid == 0) bsum[blockIdx.x] = s[0];
}

__global__ void k_loss(const u32* __restrict__ keys, const u8* __restrict__ vals,
                       const u32* __restrict__ bscan, const float* __restrict__ gtsArr,
                       float* __restrict__ loss_acc, int b){
  __shared__ u32 s[256];
  __shared__ float red[256];
  int tid = threadIdx.x;
  int base = blockIdx.x*BLKE + tid*16;
  u8 lv[16];
  u32 tsum = 0;
  #pragma unroll
  for (int j = 0; j < 16; ++j){ lv[j] = vals[base + j]; tsum += lv[j]; }
  s[tid] = tsum; __syncthreads();
  for (int off = 1; off < 256; off <<= 1){
    u32 xv = (tid >= off) ? s[tid - off] : 0u;
    __syncthreads();
    s[tid] += xv;
    __syncthreads();
  }
  u32 cprev = bscan[blockIdx.x] + (s[tid] - tsum);  // labels strictly before first elem
  float gts = gtsArr[b];
  float acc = 0.f;
  #pragma unroll
  for (int j = 0; j < 16; ++j){
    u32 lab = lv[j];
    u32 cI  = cprev + lab;              // inclusive cumsum at i
    int i   = base + j;
    float e  = dec_key(keys[i]);
    float re = fmaxf(e, 0.f);
    float fi = (float)i;
    float cf = (float)cI;
    float Ji = 1.f - (gts - cf) / (gts + (fi + 1.f) - cf);
    float Jp = 0.f;
    if (i > 0){
      float cp = (float)(cI - lab);
      Jp = 1.f - (gts - cp) / (gts + fi - cp);
    }
    acc += re * (Ji - Jp);
    cprev = cI;
  }
  red[tid] = acc; __syncthreads();
  for (int off = 128; off > 0; off >>= 1){
    if (tid < off) red[tid] += red[tid + off];
    __syncthreads();
  }
  if (tid == 0) atomicAdd(&loss_acc[b], red[0]);
}

__global__ void k_final(const float* __restrict__ loss_acc, float* __restrict__ out){
  if (threadIdx.x == 0 && blockIdx.x == 0)
    out[0] = 0.25f*(loss_acc[0] + loss_acc[1] + loss_acc[2] + loss_acc[3]);
}

extern "C" void kernel_launch(void* const* d_in, const int* in_sizes, int n_in,
                              void* d_out, int out_size, void* d_ws, size_t ws_size,
                              hipStream_t stream){
  const float* emb = (const float*)d_in[0];
  const float* w   = (const float*)d_in[1];
  const int*   gt  = (const int*)d_in[2];
  const float* W1  = (const float*)d_in[3];
  const float* b1  = (const float*)d_in[4];
  const float* W2  = (const float*)d_in[5];
  const float* b2  = (const float*)d_in[6];
  float* out = (float*)d_out;
  (void)in_sizes; (void)n_in; (void)out_size; (void)ws_size;

  char* ws = (char*)d_ws;
  size_t o = 0;
  auto alloc = [&](size_t bytes)->char*{
    o = (o + 255) & ~(size_t)255;
    char* r = ws + o;
    o += bytes;
    return r;
  };
  float* loss_acc    = (float*)alloc(4*sizeof(float));
  float* gtsArr      = (float*)alloc(4*sizeof(float));
  size_t zero_end    = o;  // only loss_acc + gts need zeroing
  float* counts_acc  = (float*)alloc(4*NINST*sizeof(float));
  float* centers_acc = (float*)alloc(4*NINST*C_IN*sizeof(float));
  float* centers     = (float*)alloc(4*NINST*C_IN*sizeof(float));
  float* A1b         = (float*)alloc(4*NINST*HID*sizeof(float));
  float* E1          = (float*)alloc((size_t)PPIX*HID*sizeof(float));
  u32*   keys0       = (u32*)  alloc((size_t)PTOT*sizeof(u32));
  u32*   keys1       = (u32*)  alloc((size_t)PTOT*sizeof(u32));
  u8*    vals0       = (u8*)   alloc((size_t)PTOT);
  u8*    vals1       = (u8*)   alloc((size_t)PTOT);
  u32*   hist        = (u32*)  alloc(16*NBLK*sizeof(u32));
  u32*   bsum        = (u32*)  alloc(NBLK*sizeof(u32));

  int nz = (int)((zero_end + 3)/4);
  k_zero<<<(nz + 255)/256, 256, 0, stream>>>((float*)ws, nz);
  k_stage1<<<dim3(NINST, 4), 256, 0, stream>>>(emb, w, gt, counts_acc, centers_acc, gtsArr);
  k_fincenters<<<(4*NINST*C_IN + 255)/256, 256, 0, stream>>>(centers_acc, counts_acc, centers);
  k_a1b<<<(4*NINST*HID + 255)/256, 256, 0, stream>>>(centers, W1, b1, A1b);

  for (int b = 0; b < 4; ++b){
    k_e1_wmma<<<PPIX/16/8, 256, 0, stream>>>(emb, W1, E1, b);
    k_errors<<<PPIX/256, 256, 0, stream>>>(E1, A1b, W2, b2, gt, keys0, vals0, b);
    u32* ks = keys0; u32* kd = keys1;
    u8*  vs = vals0; u8*  vd = vals1;
    for (int pass = 0; pass < 8; ++pass){
      int shift = pass*4;
      k_hist<<<NBLK, 256, 0, stream>>>(ks, hist, shift);
      k_scan_excl<<<1, 256, 0, stream>>>(hist, 16*NBLK);
      k_scatter<<<NBLK, 256, 0, stream>>>(ks, vs, kd, vd, hist, shift);
      u32* tk = ks; ks = kd; kd = tk;
      u8*  tv = vs; vs = vd; vd = tv;
    }
    // 8 passes: sorted data back in keys0/vals0
    k_bsum<<<NBLK, 256, 0, stream>>>(vals0, bsum);
    k_scan_excl<<<1, 256, 0, stream>>>(bsum, NBLK);
    k_loss<<<NBLK, 256, 0, stream>>>(keys0, vals0, bsum, gtsArr, loss_acc, b);
  }
  k_final<<<1, 32, 0, stream>>>(loss_acc, out);
}
